// MultiheadAttention_33724083208266
// MI455X (gfx1250) — matrix operand order repaired and checked by reading, not compiled, not run on hardware
//
#include <hip/hip_runtime.h>

typedef __bf16 v16bf __attribute__((ext_vector_type(16)));
typedef float  v8f   __attribute__((ext_vector_type(8)));
typedef unsigned int u32x4 __attribute__((ext_vector_type(4)));
typedef int          i32x4 __attribute__((ext_vector_type(4)));
typedef int          i32x8 __attribute__((ext_vector_type(8)));

union FragBF { v16bf v; unsigned u[8]; };

// float -> bf16 round-to-nearest-even
__device__ __forceinline__ unsigned short f2bf(float f) {
  union { float f; unsigned u; } c; c.f = f;
  unsigned u = c.u;
  u += 0x7fffu + ((u >> 16) & 1u);
  return (unsigned short)(u >> 16);
}

// A-matrix (16x32 bf16) dword d (0..7) -> starting K for this lane-half.
// Contiguous order: lanes 0-15 hold K {0..15}, lanes 16-31 hold K {16..31}.
__device__ __forceinline__ int a_k0(int d, int hf) {
  return hf * 16 + 2 * d;
}

// ---------------------------------------------------------------------------
// TDM: load a 64x64 bf16 tile (row stride 64 elems in memory) into LDS with
// 1-dword padding every 32 dwords -> LDS row pitch 66 halfwords.
// D# per CDNA5 ISA (2-D: groups 2/3 zero).
// 6-arg builtin form (g0, g1, g2, g3, g4, cpol).
// ---------------------------------------------------------------------------
__device__ __forceinline__ void tdm_load_tile64x64_bf16(const void* gsrc,
                                                        unsigned ldsByteOff) {
  unsigned long long ga = (unsigned long long)(size_t)gsrc;
  u32x4 g0;
  g0[0] = 1u;                                           // count=1, user mode
  g0[1] = ldsByteOff;                                   // lds_addr
  g0[2] = (unsigned)(ga & 0xffffffffu);                 // global_addr[31:0]
  g0[3] = (unsigned)((ga >> 32) & 0x01ffffffu)          // global_addr[56:32]
        | (2u << 30);                                   // type=2 ("image")
  i32x8 g1;
  g1[0] = (1 << 16)                                     // data_size=2 bytes
        | (1 << 20)                                     // pad_enable
        | (4 << 22);                                    // pad_interval: 32 dwords
                                                        // pad_amount: code 0 = 1 dword
  g1[1] = (64 << 16);                                   // tensor_dim0 = 64
  g1[2] = (2048 << 16);                                 // tensor_dim1 = 2048 (lo16)
  g1[3] = (64 << 16);                                   // tile_dim0 = 64
  g1[4] = 64;                                           // tile_dim1 = 64, tile_dim2 = 0
  g1[5] = 64;                                           // tensor_dim0_stride = 64
  g1[6] = 0;                                            // tensor_dim1_stride unused (2D)
  g1[7] = 0;
  i32x4 z4 = {0, 0, 0, 0};
  i32x8 z8 = {0, 0, 0, 0, 0, 0, 0, 0};
  __builtin_amdgcn_tensor_load_to_lds(g0, g1, z4, z4, z8, 0);
}

// ---------------------------------------------------------------------------
// Generic projection GEMM: Y = X @ W^T + bias   (M x 1024) * (1024 x 1024)
// MODE 0: fp32 row-major [M][1024] (final output projection)
// MODE 1: bf16 head-major [b][h][n][64] (Q/K/V staging), value scaled
// Block: 256 thr (8 waves), 64x64 output tile, K stepped by 32.
// ---------------------------------------------------------------------------
template <int MODE>
__global__ __launch_bounds__(256)
void mha_proj_kernel(const float* __restrict__ X, const float* __restrict__ W,
                     const float* __restrict__ bias, void* __restrict__ out,
                     float scale) {
  __shared__ unsigned short sA[64][34];   // X tile [m][k], padded (17 dwords/row)
  __shared__ unsigned short sB[64][34];   // W tile [n][k] (W row n = col n of W^T)

  const int tid  = threadIdx.x;
  const int lane = tid & 31;
  const int wave = tid >> 5;
  const int hf   = lane >> 4;
  const int ln   = lane & 15;
  const int Mbase = blockIdx.y * 64;
  const int Nbase = blockIdx.x * 64;
  const int wr = (wave >> 1) * 16;
  const int wc = (wave & 1) * 32;

  v8f acc0, acc1;
  #pragma unroll
  for (int i = 0; i < 8; ++i) { acc0[i] = 0.f; acc1[i] = 0.f; }

  for (int k0 = 0; k0 < 1024; k0 += 32) {
    #pragma unroll
    for (int it = 0; it < 2; ++it) {             // 512 float4 loads / 256 thr
      int i = tid + it * 256;
      int row = i >> 3, cg = i & 7;
      float4 fx = *(const float4*)&X[(size_t)(Mbase + row) * 1024 + k0 + cg * 4];
      unsigned short* da = &sA[row][cg * 4];
      da[0] = f2bf(fx.x); da[1] = f2bf(fx.y); da[2] = f2bf(fx.z); da[3] = f2bf(fx.w);
      float4 fw = *(const float4*)&W[(size_t)(Nbase + row) * 1024 + k0 + cg * 4];
      unsigned short* db = &sB[row][cg * 4];
      db[0] = f2bf(fw.x); db[1] = f2bf(fw.y); db[2] = f2bf(fw.z); db[3] = f2bf(fw.w);
    }
    __syncthreads();

    FragBF a, b0, b1;
    #pragma unroll
    for (int d = 0; d < 8; ++d) {
      a.u[d]  = *(const unsigned*)&sA[wr + ln][a_k0(d, hf)];
      b0.u[d] = *(const unsigned*)&sB[wc + ln][hf * 16 + 2 * d];       // B: K = hf*16+e
      b1.u[d] = *(const unsigned*)&sB[wc + 16 + ln][hf * 16 + 2 * d];
    }
    acc0 = __builtin_amdgcn_wmma_f32_16x16x32_bf16(false, a.v, false, b0.v,
                                                   (short)0, acc0, false, false);
    acc1 = __builtin_amdgcn_wmma_f32_16x16x32_bf16(false, a.v, false, b1.v,
                                                   (short)0, acc1, false, false);
    __syncthreads();
  }

  #pragma unroll
  for (int nt = 0; nt < 2; ++nt) {
    v8f acc = nt ? acc1 : acc0;
    int col = Nbase + wc + nt * 16 + ln;
    float bv = bias[col];
    #pragma unroll
    for (int r = 0; r < 8; ++r) {
      int m = Mbase + wr + r + hf * 8;           // C layout: VGPR r -> row r / r+8
      float val = (acc[r] + bv) * scale;
      if (MODE == 0) {
        ((float*)out)[(size_t)m * 1024 + col] = val;
      } else {
        int bb = m >> 11, ns = m & 2047, hh = col >> 6, dd = col & 63;
        ((unsigned short*)out)[((size_t)(bb * 16 + hh) * 2048 + ns) * 64 + dd] = f2bf(val);
      }
    }
  }
}

// ---------------------------------------------------------------------------
// Flash attention: grid (16 q-tiles of 128 rows, 32 bh). 8 waves x 16 q-rows.
// Q pre-scaled by 1/sqrt(d). K tile staged by the Tensor Data Mover; V tile
// staged transposed by the waves. Streams 64-key tiles through LDS.
// ---------------------------------------------------------------------------
__global__ __launch_bounds__(256)
void mha_attn_kernel(const unsigned short* __restrict__ qh,
                     const unsigned short* __restrict__ kh,
                     const unsigned short* __restrict__ vh,
                     float* __restrict__ attnOut) {
  __shared__ unsigned short sK [64][66];     // K tile  [key][d]  (TDM-filled)
  __shared__ unsigned short sVT[64][66];     // V tile transposed [d][key]
  __shared__ unsigned short sP [8][16][66];  // per-wave P staging [row][key]

  const int tid  = threadIdx.x;
  const int lane = tid & 31;
  const int wave = tid >> 5;
  const int hf   = lane >> 4;
  const int ln   = lane & 15;
  const int bh = blockIdx.y;                 // 0..31
  const int bb = bh >> 4, hh = bh & 15;
  const int qbase = blockIdx.x * 128;
  const int qrow  = qbase + wave * 16 + ln;

  const unsigned* qh32 = (const unsigned*)qh;
  const unsigned* vh32 = (const unsigned*)vh;
  const size_t bhRow = (size_t)bh * 2048;
  const unsigned sKlds = (unsigned)(size_t)&sK[0][0];   // LDS byte offset (addr[31:0])

  // Q fragments (A-layout), two k=32 chunks covering d=0..63
  FragBF qa[2];
  {
    size_t base32 = (bhRow + qrow) * 32;     // 32 dwords per 64-bf16 row
    #pragma unroll
    for (int c = 0; c < 2; ++c)
      #pragma unroll
      for (int d = 0; d < 8; ++d)
        qa[c].u[d] = qh32[base32 + (unsigned)(32 * c + a_k0(d, hf)) / 2u];
  }

  float mrow[8], lrow[8];
  v8f oacc[4];
  #pragma unroll
  for (int r = 0; r < 8; ++r) { mrow[r] = -1e30f; lrow[r] = 0.f; }
  #pragma unroll
  for (int a = 0; a < 4; ++a)
    #pragma unroll
    for (int i = 0; i < 8; ++i) oacc[a][i] = 0.f;

  const float LOG2E = 1.44269504088896f;

  for (int kt = 0; kt < 32; ++kt) {
    // TDM pulls the K tile into LDS (wave 0 issues; EXEC-independent DMA).
    if (wave == 0)
      tdm_load_tile64x64_bf16(kh + (bhRow + (size_t)kt * 64) * 64, sKlds);

    // Meanwhile all waves stage the V tile transposed (2048 dwords).
    #pragma unroll
    for (int it = 0; it < 8; ++it) {
      int i = tid + it * 256;
      int row = i >> 5, cg = i & 31;
      unsigned vv = vh32[(bhRow + (size_t)kt * 64 + row) * 32 + cg];
      sVT[cg * 2 + 0][row] = (unsigned short)(vv & 0xffffu);
      sVT[cg * 2 + 1][row] = (unsigned short)(vv >> 16);
    }

    // Prefetch next tile's K/V lines into GL2 while we compute this one.
    if (kt + 1 < 32) {
      const unsigned short* pk = &kh[(bhRow + (size_t)(kt + 1) * 64 + (tid & 63)) * 64];
      const unsigned short* pv = &vh[(bhRow + (size_t)(kt + 1) * 64 + (tid & 63)) * 64];
      __builtin_prefetch(pk, 0, 1);
      __builtin_prefetch(pv, 0, 1);
    }

    if (wave == 0) __builtin_amdgcn_s_wait_tensorcnt(0);
    __syncthreads();

    // S = Q @ K^T  (16 x 64 per wave, 8 WMMAs)
    v8f sacc[4];
    #pragma unroll
    for (int a = 0; a < 4; ++a)
      #pragma unroll
      for (int i = 0; i < 8; ++i) sacc[a][i] = 0.f;

    #pragma unroll
    for (int c = 0; c < 2; ++c)
      #pragma unroll
      for (int a = 0; a < 4; ++a) {
        FragBF b;   // B[k=d][n=key]: fixed key row in sK, d contiguous
        #pragma unroll
        for (int d = 0; d < 8; ++d)
          b.u[d] = *(const unsigned*)&sK[a * 16 + ln][32 * c + hf * 16 + 2 * d];
        sacc[a] = __builtin_amdgcn_wmma_f32_16x16x32_bf16(false, qa[c].v, false, b.v,
                                                          (short)0, sacc[a], false, false);
      }

    // online softmax (per row r; lane-half reductions stay inside 16 lanes)
    #pragma unroll
    for (int r = 0; r < 8; ++r) {
      float mx = fmaxf(fmaxf(sacc[0][r], sacc[1][r]), fmaxf(sacc[2][r], sacc[3][r]));
      mx = fmaxf(mx, __shfl_xor(mx, 1, 32));
      mx = fmaxf(mx, __shfl_xor(mx, 2, 32));
      mx = fmaxf(mx, __shfl_xor(mx, 4, 32));
      mx = fmaxf(mx, __shfl_xor(mx, 8, 32));
      float mnew  = fmaxf(mrow[r], mx);
      float alpha = exp2f((mrow[r] - mnew) * LOG2E);
      float rs = 0.f;
      #pragma unroll
      for (int a = 0; a < 4; ++a) {
        float p = exp2f((sacc[a][r] - mnew) * LOG2E);
        sacc[a][r] = p;
        rs += p;
      }
      rs += __shfl_xor(rs, 1, 32);
      rs += __shfl_xor(rs, 2, 32);
      rs += __shfl_xor(rs, 4, 32);
      rs += __shfl_xor(rs, 8, 32);
      lrow[r] = lrow[r] * alpha + rs;
      mrow[r] = mnew;
      #pragma unroll
      for (int a = 0; a < 4; ++a) oacc[a][r] *= alpha;
    }

    // stage P (bf16) in this wave's private strip (same-wave DS ordering)
    #pragma unroll
    for (int a = 0; a < 4; ++a)
      #pragma unroll
      for (int r = 0; r < 8; ++r)
        sP[wave][r + hf * 8][a * 16 + ln] = f2bf(sacc[a][r]);

    // O += P @ V  (8 WMMAs)
    #pragma unroll
    for (int c = 0; c < 2; ++c) {
      FragBF pa;
      #pragma unroll
      for (int d = 0; d < 8; ++d)
        pa.u[d] = *(const unsigned*)&sP[wave][ln][32 * c + a_k0(d, hf)];
      #pragma unroll
      for (int a = 0; a < 4; ++a) {
        FragBF b;   // B[k=key][n=d]: fixed d row in sVT, key contiguous
        #pragma unroll
        for (int d = 0; d < 8; ++d)
          b.u[d] = *(const unsigned*)&sVT[a * 16 + ln][32 * c + hf * 16 + 2 * d];
        oacc[a] = __builtin_amdgcn_wmma_f32_16x16x32_bf16(false, pa.v, false, b.v,
                                                          (short)0, oacc[a], false, false);
      }
    }
    __syncthreads();   // all reads of sK/sVT done before next TDM/staging
  }

  // normalize; store fp32 [b, n, c] for the output projection
  #pragma unroll
  for (int a = 0; a < 4; ++a) {
    int col = hh * 64 + a * 16 + ln;
    #pragma unroll
    for (int r = 0; r < 8; ++r) {
      int row = qbase + wave * 16 + r + hf * 8;
      attnOut[((size_t)bb * 2048 + row) * 1024 + col] = oacc[a][r] / lrow[r];
    }
  }
}

// ---------------------------------------------------------------------------
extern "C" void kernel_launch(void* const* d_in, const int* in_sizes, int n_in,
                              void* d_out, int out_size, void* d_ws, size_t ws_size,
                              hipStream_t stream) {
  const float* q  = (const float*)d_in[0];
  const float* k  = (const float*)d_in[1];
  const float* v  = (const float*)d_in[2];
  const float* wq = (const float*)d_in[3];
  const float* bq = (const float*)d_in[4];
  const float* wk = (const float*)d_in[5];
  const float* bk = (const float*)d_in[6];
  const float* wv = (const float*)d_in[7];
  const float* bv = (const float*)d_in[8];
  const float* wo = (const float*)d_in[9];
  const float* bo = (const float*)d_in[10];

  // workspace: qh|kh|vh bf16 head-major (8 MB each) + attn fp32 (16 MB)
  const size_t ELEMS = (size_t)2 * 16 * 2048 * 64;   // 4,194,304
  unsigned short* qh = (unsigned short*)d_ws;
  unsigned short* kh = qh + ELEMS;
  unsigned short* vh = kh + ELEMS;
  float* attn = (float*)(vh + ELEMS);

  dim3 blk(256);
  dim3 pg(16, 64);                 // 1024/64 N-tiles x 4096/64 M-tiles
  const float SCALE = 0.125f;      // 64^-0.5, folded into Q

  mha_proj_kernel<1><<<pg, blk, 0, stream>>>(q, wq, bq, (void*)qh, SCALE);
  mha_proj_kernel<1><<<pg, blk, 0, stream>>>(k, wk, bk, (void*)kh, 1.0f);
  mha_proj_kernel<1><<<pg, blk, 0, stream>>>(v, wv, bv, (void*)vh, 1.0f);
  mha_attn_kernel<<<dim3(16, 32), blk, 0, stream>>>(qh, kh, vh, attn);
  mha_proj_kernel<0><<<pg, blk, 0, stream>>>(attn, wo, bo, d_out, 1.0f);
}
